// TransformerLayer_9912784519715
// MI455X (gfx1250) — matrix operand order repaired and checked by reading, not compiled, not run on hardware
//
#include <hip/hip_runtime.h>

typedef __attribute__((ext_vector_type(16))) _Float16 v16h;
typedef __attribute__((ext_vector_type(8)))  float    v8f;
typedef unsigned int u32;

#define H_DIM 512
#define NHEAD 8
#define HD 64

// ---------------------------------------------------------------------------
// A-fragment loader: 16x32 f16 A tile in contiguous-K order (B loads here use the same order).
// lane: M = lane&15, hi = lane>>4. VGPR v holds f16 pair at
//   K = hi*16 + v*2  (+0/+1)
// rowPtr points at element (M, 0) of the 16x32 tile (row-major, contiguous K).
// Works for global or LDS pointers.
// ---------------------------------------------------------------------------
__device__ __forceinline__ v16h load_a_frag(const _Float16* rowPtr, int hi) {
    v16h a;
    u32* au = reinterpret_cast<u32*>(&a);
#pragma unroll
    for (int v = 0; v < 8; ++v) {
        int koff = hi * 16 + v * 2;
        au[v] = *reinterpret_cast<const u32*>(rowPtr + koff);
    }
    return a;
}

// ---------------------------------------------------------------------------
// Tiled WMMA GEMM:  out = act(A[M x K] * Bt^T + bias)
//   A  : f16 row-major [Mpad x K]   (Mpad = gridDim.y*256, zero padded)
//   Bt : f16, B transposed, [N x K] row-major -> contiguous K per column n
// block = 256 thr = 8 waves; wave w -> 32 rows starting blockIdx.y*256 + w*32,
// cols = blockIdx.x*64. Per k-step: 2 A frags + 4 B frags -> 8 WMMAs
// (each B fragment reused twice -> half the B-stream bytes per FLOP).
// GELU: exact erf gelu. OUT16: write f16 [ld=N]; else f32 guarded by m<Mstore.
// ---------------------------------------------------------------------------
template <int GELU, int OUT16>
__global__ void gemm_wmma(const _Float16* __restrict__ A,
                          const _Float16* __restrict__ Bt,
                          const float* __restrict__ bias,
                          _Float16* __restrict__ out16,
                          float* __restrict__ out32,
                          int Mstore, int N, int K) {
    const int wave = threadIdx.x >> 5;
    const int lane = threadIdx.x & 31;
    const int lo = lane & 15, hi = lane >> 4;
    const int row = blockIdx.y * 256 + wave * 32;
    const int col = blockIdx.x * 64;

    v8f c[2][4] = {{{}, {}, {}, {}}, {{}, {}, {}, {}}};
    const _Float16* arow0 = A + (size_t)(row + lo) * K;
    const _Float16* arow1 = arow0 + (size_t)16 * K;
    const _Float16* bbase = Bt + (size_t)(col + lo) * K + hi * 16;

    for (int k0 = 0; k0 < K; k0 += 32) {
        v16h a0 = load_a_frag(arow0 + k0, hi);
        v16h a1 = load_a_frag(arow1 + k0, hi);
#pragma unroll
        for (int j = 0; j < 4; ++j) {
            v16h b = *reinterpret_cast<const v16h*>(bbase + (size_t)(j * 16) * K + k0);
            c[0][j] = __builtin_amdgcn_wmma_f32_16x16x32_f16(false, a0, false, b,
                                                             (short)0, c[0][j], false, false);
            c[1][j] = __builtin_amdgcn_wmma_f32_16x16x32_f16(false, a1, false, b,
                                                             (short)0, c[1][j], false, false);
        }
    }

#pragma unroll
    for (int j = 0; j < 4; ++j) {
        int n = col + j * 16 + lo;
        float bcol = bias[n];
#pragma unroll
        for (int i = 0; i < 2; ++i) {
#pragma unroll
            for (int r = 0; r < 8; ++r) {
                int m = row + i * 16 + r + hi * 8;
                float val = c[i][j][r] + bcol;
                if (GELU) val = 0.5f * val * (1.0f + erff(val * 0.70710678118654752f));
                if (OUT16) {
                    out16[(size_t)m * N + n] = (_Float16)val;
                } else if (m < Mstore) {
                    out32[(size_t)m * N + n] = val;
                }
            }
        }
    }
}

// ---------------------------------------------------------------------------
// Ragged flash attention. grid = B*NH blocks, 256 thr (8 waves).
// qkv layout [Tpad x 1536]: cols 0..511 = V, 512..1023 = Q, 1024..1535 = K.
// vT layout [512 x Tpad] = V transposed (head-major columns).
// Each wave: 16-query tile, streams 32-key tiles, online softmax.
// ---------------------------------------------------------------------------
__global__ void attn_wmma(const _Float16* __restrict__ qkv,
                          const _Float16* __restrict__ vT,
                          _Float16* __restrict__ outA,
                          const int* __restrict__ cum,
                          int Tpad) {
    __shared__ _Float16 pbuf[8 * 16 * 32];  // per-wave 16x32 P staging (8 KB)

    const int b = blockIdx.x >> 3;
    const int h = blockIdx.x & 7;
    const int start = cum[b];
    const int n = cum[b + 1] - start;
    const int wave = threadIdx.x >> 5;
    const int lane = threadIdx.x & 31;
    const int lo = lane & 15, hi = lane >> 4;
    _Float16* pb = pbuf + wave * 512;

    const int nQT = (n + 15) >> 4;
    for (int qt = wave; qt < nQT; qt += 8) {
        const int qRow = start + qt * 16;
        const _Float16* qptr = qkv + (size_t)(qRow + lo) * 1536 + 512 + h * 64;
        v16h aq0 = load_a_frag(qptr, hi);
        v16h aq1 = load_a_frag(qptr + 32, hi);

        v8f o[4] = {{}, {}, {}, {}};
        float mrow[8], lrow[8];
#pragma unroll
        for (int r = 0; r < 8; ++r) { mrow[r] = -3.0e38f; lrow[r] = 0.0f; }

        for (int kb = 0; kb < n; kb += 32) {
            // ---- S = Q * K^T (two 16x16 tiles: keys kb..kb+15 and kb+16..kb+31)
            v8f s0 = {}, s1 = {};
            const _Float16* kp0 =
                qkv + (size_t)(start + kb + lo) * 1536 + 1024 + h * 64 + hi * 16;
            const _Float16* kp1 = kp0 + (size_t)16 * 1536;
            v16h bk;
            bk = *reinterpret_cast<const v16h*>(kp0);
            s0 = __builtin_amdgcn_wmma_f32_16x16x32_f16(false, aq0, false, bk, (short)0, s0, false, false);
            bk = *reinterpret_cast<const v16h*>(kp0 + 32);
            s0 = __builtin_amdgcn_wmma_f32_16x16x32_f16(false, aq1, false, bk, (short)0, s0, false, false);
            bk = *reinterpret_cast<const v16h*>(kp1);
            s1 = __builtin_amdgcn_wmma_f32_16x16x32_f16(false, aq0, false, bk, (short)0, s1, false, false);
            bk = *reinterpret_cast<const v16h*>(kp1 + 32);
            s1 = __builtin_amdgcn_wmma_f32_16x16x32_f16(false, aq1, false, bk, (short)0, s1, false, false);

            // ---- scale + key-validity mask
            const bool v0 = (kb + lo) < n;
            const bool v1 = (kb + 16 + lo) < n;
#pragma unroll
            for (int r = 0; r < 8; ++r) {
                s0[r] = v0 ? s0[r] * 0.125f : -1.0e9f;
                s1[r] = v1 ? s1[r] * 0.125f : -1.0e9f;
            }

            // ---- row max across the 16 lanes of each half-wave
            float t[8];
#pragma unroll
            for (int r = 0; r < 8; ++r) t[r] = fmaxf(s0[r], s1[r]);
#pragma unroll
            for (int d = 1; d < 16; d <<= 1)
#pragma unroll
                for (int r = 0; r < 8; ++r) t[r] = fmaxf(t[r], __shfl_xor(t[r], d, 32));

            // ---- online softmax update
            float p0[8], p1[8], sum[8], corr[8];
#pragma unroll
            for (int r = 0; r < 8; ++r) {
                float mn = fmaxf(mrow[r], t[r]);
                corr[r] = expf(mrow[r] - mn);
                mrow[r] = mn;
                p0[r] = expf(s0[r] - mn);
                p1[r] = expf(s1[r] - mn);
                sum[r] = p0[r] + p1[r];
            }
#pragma unroll
            for (int d = 1; d < 16; d <<= 1)
#pragma unroll
                for (int r = 0; r < 8; ++r) sum[r] += __shfl_xor(sum[r], d, 32);
#pragma unroll
            for (int r = 0; r < 8; ++r) {
                lrow[r] = lrow[r] * corr[r] + sum[r];
#pragma unroll
                for (int j = 0; j < 4; ++j) o[j][r] *= corr[r];
            }

            // ---- transpose P (C-layout -> A-layout) through LDS
#pragma unroll
            for (int r = 0; r < 8; ++r) {
                pb[(r + hi * 8) * 32 + lo] = (_Float16)p0[r];
                pb[(r + hi * 8) * 32 + 16 + lo] = (_Float16)p1[r];
            }
            asm volatile("" ::: "memory");  // order LDS write -> LDS read (in-order DS pipe)
            v16h pa = load_a_frag(pb + lo * 32, hi);
            asm volatile("" ::: "memory");  // order LDS read -> next-iter LDS write

            // ---- O += P * V  (V^T is [hd][token], unaligned in token -> scalar loads)
#pragma unroll
            for (int j = 0; j < 4; ++j) {
                const _Float16* vp =
                    vT + (size_t)(h * 64 + j * 16 + lo) * Tpad + start + kb + hi * 16;
                v16h bv;
#pragma unroll
                for (int s = 0; s < 16; ++s) bv[s] = vp[s];
                o[j] = __builtin_amdgcn_wmma_f32_16x16x32_f16(false, pa, false, bv,
                                                              (short)0, o[j], false, false);
            }
        }

        // ---- normalize and store
#pragma unroll
        for (int j = 0; j < 4; ++j)
#pragma unroll
            for (int r = 0; r < 8; ++r) {
                int q = qt * 16 + r + hi * 8;
                if (q < n)
                    outA[(size_t)(start + q) * 512 + h * 64 + j * 16 + lo] =
                        (_Float16)(o[j][r] / lrow[r]);
            }
    }
}

// ---------------------------------------------------------------------------
// LayerNorm(x1 + x2) * g + beta. One block per token, 256 thr x 2 cols.
// Optionally writes an f16 copy (outH). grid may cover Tpad: rows >= T only
// zero-fill outH (padding rows for downstream WMMA GEMMs).
// ---------------------------------------------------------------------------
__global__ void ln_resid(const float* __restrict__ x1, const float* __restrict__ x2,
                         const float* __restrict__ g, const float* __restrict__ beta,
                         float* __restrict__ outF, _Float16* __restrict__ outH, int T) {
    const int t = blockIdx.x;
    const int tid = threadIdx.x;
    if (t >= T) {
        if (outH) {
            outH[(size_t)t * 512 + tid] = (_Float16)0.0f;
            outH[(size_t)t * 512 + tid + 256] = (_Float16)0.0f;
        }
        return;
    }
    const size_t base = (size_t)t * 512;
    float a0 = x1[base + tid] + x2[base + tid];
    float a1 = x1[base + tid + 256] + x2[base + tid + 256];

    __shared__ float s1[256], s2[256];
    s1[tid] = a0 + a1;
    s2[tid] = a0 * a0 + a1 * a1;
    __syncthreads();
    for (int st = 128; st > 0; st >>= 1) {
        if (tid < st) { s1[tid] += s1[tid + st]; s2[tid] += s2[tid + st]; }
        __syncthreads();
    }
    const float mu = s1[0] * (1.0f / 512.0f);
    const float var = s2[0] * (1.0f / 512.0f) - mu * mu;
    const float rs = rsqrtf(var + 1e-5f);
    float y0 = (a0 - mu) * rs * g[tid] + beta[tid];
    float y1 = (a1 - mu) * rs * g[tid + 256] + beta[tid + 256];
    if (outF) { outF[base + tid] = y0; outF[base + tid + 256] = y1; }
    if (outH) { outH[base + tid] = (_Float16)y0; outH[base + tid + 256] = (_Float16)y1; }
}

// --------------------------- conversion kernels ----------------------------
__global__ void cvt_h16(const float* __restrict__ src, _Float16* __restrict__ dst,
                        long long nvalid, long long ntotal) {
    long long i = (long long)blockIdx.x * 256 + threadIdx.x;
    if (i < ntotal) dst[i] = (i < nvalid) ? (_Float16)src[i] : (_Float16)0.0f;
}

// Wqkv^T [1536 x 512]: rows 0..511 from Wv, rows 512..1535 from Wqk columns.
__global__ void cvt_wqkv(const float* __restrict__ Wv, const float* __restrict__ Wqk,
                         _Float16* __restrict__ Wt) {
    int i = blockIdx.x * 256 + threadIdx.x;  // 1536*512
    if (i >= 1536 * 512) return;
    int nn = i >> 9, kk = i & 511;
    float v = (nn < 512) ? Wv[kk * 512 + nn] : Wqk[kk * 1024 + (nn - 512)];
    Wt[i] = (_Float16)v;
}

__global__ void cvt_wT(const float* __restrict__ W, _Float16* __restrict__ Wt,
                       int K, int N) {
    long long i = (long long)blockIdx.x * 256 + threadIdx.x;
    if (i >= (long long)K * N) return;
    int k = (int)(i / N), nn = (int)(i % N);
    Wt[(size_t)nn * K + k] = (_Float16)W[i];
}

__global__ void cvt_bias_qkv(const float* __restrict__ bv, const float* __restrict__ bqk,
                             float* __restrict__ dst) {
    int i = blockIdx.x * 256 + threadIdx.x;
    if (i < 1536) dst[i] = (i < 512) ? bv[i] : bqk[i - 512];
}

// vT[c][t] = qkv[t][c] for c in [0,512) (the V block).
__global__ void transpose_v(const _Float16* __restrict__ qkv, _Float16* __restrict__ vT,
                            int Tpad) {
    long long i = (long long)blockIdx.x * 256 + threadIdx.x;
    if (i >= (long long)512 * Tpad) return;
    int c = (int)(i / Tpad), t = (int)(i % Tpad);
    vT[i] = qkv[(size_t)t * 1536 + c];
}

// ---------------------------------------------------------------------------
extern "C" void kernel_launch(void* const* d_in, const int* in_sizes, int n_in,
                              void* d_out, int out_size, void* d_ws, size_t ws_size,
                              hipStream_t stream) {
    const float* h_n = (const float*)d_in[0];
    const float* Wv  = (const float*)d_in[1];
    const float* bv  = (const float*)d_in[2];
    const float* Wqk = (const float*)d_in[3];
    const float* bqk = (const float*)d_in[4];
    const float* Wp  = (const float*)d_in[5];
    const float* bp  = (const float*)d_in[6];
    const float* g1  = (const float*)d_in[7];
    const float* b1  = (const float*)d_in[8];
    const float* W1  = (const float*)d_in[9];
    const float* bf1 = (const float*)d_in[10];
    const float* W2  = (const float*)d_in[11];
    const float* bf2 = (const float*)d_in[12];
    const float* g2  = (const float*)d_in[13];
    const float* b2  = (const float*)d_in[14];
    const int*   cum = (const int*)d_in[15];

    const int T = in_sizes[0] / H_DIM;
    // padded rows: multiple of 256 (GEMM M-tile) and >= T+31 (attention overread)
    const int Tpad = ((T + 31 + 255) / 256) * 256;
    const int nb = in_sizes[15] - 1;  // number of sequences (B)

    char* ws = (char*)d_ws;
    size_t off = 0;
    auto alloc = [&](size_t bytes) {
        size_t o = off;
        off += (bytes + 255) & ~(size_t)255;
        return o;
    };
    _Float16* WqkvT = (_Float16*)(ws + alloc((size_t)1536 * 512 * 2));
    _Float16* WpT   = (_Float16*)(ws + alloc((size_t)512 * 512 * 2));
    _Float16* W1T   = (_Float16*)(ws + alloc((size_t)2048 * 512 * 2));
    _Float16* W2T   = (_Float16*)(ws + alloc((size_t)512 * 2048 * 2));
    float*    bqkv  = (float*)(ws + alloc((size_t)1536 * 4));
    _Float16* h16   = (_Float16*)(ws + alloc((size_t)Tpad * 512 * 2));  // later: vT
    _Float16* qkv16 = (_Float16*)(ws + alloc((size_t)Tpad * 1536 * 2));
    _Float16* attn16= (_Float16*)(ws + alloc((size_t)Tpad * 512 * 2));
    float*    f32tmp= (float*)(ws + alloc((size_t)T * 512 * 4));        // pout, later ff2
    float*    hln   = (float*)(ws + alloc((size_t)T * 512 * 4));
    _Float16* hln16 = (_Float16*)(ws + alloc((size_t)Tpad * 512 * 2));
    _Float16* ff1   = (_Float16*)(ws + alloc((size_t)Tpad * 2048 * 2));
    _Float16* vT    = h16;  // alias: h16 dead after QKV GEMM

    // 1) conversions
    cvt_wqkv<<<(1536 * 512 + 255) / 256, 256, 0, stream>>>(Wv, Wqk, WqkvT);
    cvt_wT<<<(512 * 512 + 255) / 256, 256, 0, stream>>>(Wp, WpT, 512, 512);
    cvt_wT<<<(512 * 2048 + 255) / 256, 256, 0, stream>>>(W1, W1T, 512, 2048);
    cvt_wT<<<(2048 * 512 + 255) / 256, 256, 0, stream>>>(W2, W2T, 2048, 512);
    cvt_bias_qkv<<<6, 256, 0, stream>>>(bv, bqk, bqkv);
    {
        long long tot = (long long)Tpad * 512;
        cvt_h16<<<(int)((tot + 255) / 256), 256, 0, stream>>>(h_n, h16,
                                                              (long long)T * 512, tot);
    }

    // 2) fused QKV projection: qkv16 = h16 @ [Wv|Wqk] + [bv|bqk]  (f16 out)
    gemm_wmma<0, 1><<<dim3(1536 / 64, Tpad / 256), 256, 0, stream>>>(
        h16, WqkvT, bqkv, qkv16, nullptr, T, 1536, 512);

    // 3) V transpose for attention P*V fragments
    {
        long long tot = (long long)512 * Tpad;
        transpose_v<<<(int)((tot + 255) / 256), 256, 0, stream>>>(qkv16, vT, Tpad);
    }

    // 4) ragged flash attention
    attn_wmma<<<nb * NHEAD, 256, 0, stream>>>(qkv16, vT, attn16, cum, Tpad);

    // 5) output projection (f32)
    gemm_wmma<0, 0><<<dim3(512 / 64, Tpad / 256), 256, 0, stream>>>(
        attn16, WpT, bp, nullptr, f32tmp, T, 512, 512);

    // 6) residual + LN1 (writes f32 hln and f16 hln16 with zero padding)
    ln_resid<<<Tpad, 256, 0, stream>>>(h_n, f32tmp, g1, b1, hln, hln16, T);

    // 7) FFN up-projection + exact GELU (f16 out)
    gemm_wmma<1, 1><<<dim3(2048 / 64, Tpad / 256), 256, 0, stream>>>(
        hln16, W1T, bf1, ff1, nullptr, T, 2048, 512);

    // 8) FFN down-projection (f32, reuse f32tmp)
    gemm_wmma<0, 0><<<dim3(512 / 64, Tpad / 256), 256, 0, stream>>>(
        ff1, W2T, bf2, nullptr, f32tmp, T, 512, 2048);

    // 9) residual + LN2 -> d_out
    ln_resid<<<T, 256, 0, stream>>>(hln, f32tmp, g2, b2, (float*)d_out, nullptr, T);
}
